// SlidingWindowAttention_73040213836405
// MI455X (gfx1250) — matrix operand order repaired and checked by reading, not compiled, not run on hardware
//
#include <hip/hip_runtime.h>

typedef __attribute__((ext_vector_type(16))) __bf16 v16bf;
typedef __attribute__((ext_vector_type(8)))  __bf16 v8bf;
typedef __attribute__((ext_vector_type(4)))  __bf16 v4bf;
typedef __attribute__((ext_vector_type(8)))  float  v8f;
typedef __attribute__((ext_vector_type(4)))  unsigned int v4u;
typedef __attribute__((ext_vector_type(4)))  int v4i;
typedef __attribute__((ext_vector_type(8)))  int v8i;

#define WMMA_BF16(a, b, c) \
  __builtin_amdgcn_wmma_f32_16x16x32_bf16(false, (a), false, (b), (short)0, (c), false, false)

// LDS tile geometry: 64 rows x 64 bf16, TDM-padded to 144B row stride (72 elems)
#define LDS_STRIDE 72

// ---------------------------------------------------------------------------
// A-fragment loader for 16x32 bf16 A-matrix (CDNA5 layout):
// element e -> k = (e/8)*16 + 8*hi + (e%8); two contiguous 16B runs per lane.
// ---------------------------------------------------------------------------
__device__ __forceinline__ v16bf load_a_frag(const __bf16* row, int k0, int hi) {
  v8bf lo = *(const v8bf*)(row + k0 + 8 * hi);
  v8bf hh = *(const v8bf*)(row + k0 + 16 + 8 * hi);
  return __builtin_shufflevector(lo, hh, 0, 1, 2, 3, 4, 5, 6, 7, 8, 9, 10, 11, 12, 13, 14, 15);
}

// contiguous 16-element bf16 load as two 16B-aligned halves (LDS-safe)
__device__ __forceinline__ v16bf load16(const __bf16* p) {
  v8bf a = *(const v8bf*)p;
  v8bf b = *(const v8bf*)(p + 8);
  return __builtin_shufflevector(a, b, 0, 1, 2, 3, 4, 5, 6, 7, 8, 9, 10, 11, 12, 13, 14, 15);
}

__device__ __forceinline__ unsigned lds_off32(const void* p) {
  // flat LDS address: low 32 bits are the wave-relative LDS byte offset
  return (unsigned)(unsigned long long)(uintptr_t)p;
}

// ---------------------------------------------------------------------------
// TDM: async 2D tile load (64 rows x 64 bf16) global -> LDS, row stride = K=512.
// D# per CDNA5 ISA: count=1, type=2 (image),
// data_size=2B, pad every 32 DWORDs by 4 DWORDs (-> 144B LDS row stride).
// ---------------------------------------------------------------------------
__device__ __forceinline__ void tdm_tile_load(const __bf16* gbase, unsigned rows_total,
                                              unsigned row0, unsigned k0, unsigned ldsoff) {
  unsigned long long ga =
      (unsigned long long)(uintptr_t)gbase + ((unsigned long long)row0 * 512ull + k0) * 2ull;
  v4u g0;
  g0[0] = 1u;                                   // count=1, no gather
  g0[1] = ldsoff;                               // lds_addr [63:32]
  g0[2] = (unsigned)(ga & 0xFFFFFFFFu);         // global_addr [95:64]
  g0[3] = (unsigned)(ga >> 32) | 0x80000000u;   // global_addr hi | type=2 [127:126]
  v8i g1;
  g1[0] = (1 << 16)     // data_size = 2 bytes
        | (1 << 20)     // pad_enable
        | (4 << 22)     // pad_interval: 2^(4+1)=32 DWORDs (= one 128B row)
        | (3 << 25);    // pad_amount: 4 DWORDs (16B) -> 144B row stride
  g1[1] = (int)(512u << 16);                          // tensor_dim0=512 (low16 @ [63:48])
  g1[2] = (int)((512u >> 16) | (rows_total << 16));   // dim0 hi | tensor_dim1 low16
  g1[3] = (int)((rows_total >> 16) | (64u << 16));    // dim1 hi | tile_dim0=64
  g1[4] = 64;                                         // tile_dim1=64, tile_dim2=0 (2D)
  g1[5] = 512;                                        // tensor_dim0_stride[31:0]
  g1[6] = 0;                                          // stride hi, dim1_stride lo
  g1[7] = 0;
  v4i g2 = {0, 0, 0, 0};
  v4i g3 = {0, 0, 0, 0};
#if __has_include(<hip/amd_detail/amd_gfx1250_TDM.h>)
  v8i g45 = {0, 0, 0, 0, 0, 0, 0, 0};
  __builtin_amdgcn_tensor_load_to_lds(g0, g1, g2, g3, g45, 0);
#else
  __builtin_amdgcn_tensor_load_to_lds(g0, g1, g2, g3, 0);
#endif
}

// ---------------------------------------------------------------------------
// Kernel 0: fp32 -> bf16 convert (vector x4)
// ---------------------------------------------------------------------------
__global__ void swa_cvt_bf16_kernel(const float* __restrict__ src, __bf16* __restrict__ dst, int n4) {
  int idx = blockIdx.x * blockDim.x + threadIdx.x;
  if (idx < n4) {
    float4 v = ((const float4*)src)[idx];
    v4bf o;
    o[0] = (__bf16)v.x; o[1] = (__bf16)v.y; o[2] = (__bf16)v.z; o[3] = (__bf16)v.w;
    ((v4bf*)dst)[idx] = o;
  }
}

// ---------------------------------------------------------------------------
// Kernel 1: QKV projection.  qkv[m][j] = sum_k x[m][k]*W[j][k] + b[j]
// Workgroup (4 waves) computes a 64x64 tile; A/B tiles staged in LDS by the
// TDM (double-buffered); wave w computes rows 16w..16w+15 x all 64 cols.
// ---------------------------------------------------------------------------
__global__ void swa_qkv_gemm_kernel(const __bf16* __restrict__ xb,
                                    const __bf16* __restrict__ wb,
                                    const float* __restrict__ bias,
                                    __bf16* __restrict__ qo,
                                    __bf16* __restrict__ ko,
                                    __bf16* __restrict__ vt) {
  __shared__ __bf16 smem[2][2][64 * LDS_STRIDE];  // [buf][0=A tile,1=B tile]
  const int lane = threadIdx.x & 31;
  const int wave = threadIdx.x >> 5;
  const int hi = lane >> 4, lo = lane & 15;
  const int tmb = blockIdx.x / 24;   // 64 row-tiles
  const int tnb = blockIdx.x % 24;   // 24 col-tiles
  const int m0 = tmb * 64;
  const int n0 = tnb * 64;

  v8f acc[4] = {};

  if (wave == 0) {
    tdm_tile_load(xb, 4096u, (unsigned)m0, 0u, lds_off32(&smem[0][0][0]));
    tdm_tile_load(wb, 1536u, (unsigned)n0, 0u, lds_off32(&smem[0][1][0]));
  }
  for (int kk = 0; kk < 8; ++kk) {   // 8 chunks of K=64
    const int cur = kk & 1;
    if (wave == 0) {
      if (kk < 7) {
        tdm_tile_load(xb, 4096u, (unsigned)m0, (unsigned)((kk + 1) * 64),
                      lds_off32(&smem[1 - cur][0][0]));
        tdm_tile_load(wb, 1536u, (unsigned)n0, (unsigned)((kk + 1) * 64),
                      lds_off32(&smem[1 - cur][1][0]));
        __builtin_amdgcn_s_wait_tensorcnt(2);   // current chunk's 2 loads done
      } else {
        __builtin_amdgcn_s_wait_tensorcnt(0);
      }
    }
    __syncthreads();
    const __bf16* arow = &smem[cur][0][(16 * wave + lo) * LDS_STRIDE];
    const __bf16* btile = &smem[cur][1][0];
#pragma unroll
    for (int kb2 = 0; kb2 < 2; ++kb2) {
      v16bf a = load16(arow + kb2 * 32 + 16 * hi);
#pragma unroll
      for (int nt = 0; nt < 4; ++nt) {
        v16bf b = load16(btile + (nt * 16 + lo) * LDS_STRIDE + kb2 * 32 + 16 * hi);
        acc[nt] = WMMA_BF16(a, b, acc[nt]);
      }
    }
    if (kk < 7) __syncthreads();
  }

  // epilogue: split into q/k (per-head [bh][n][hd]) and v (transposed [bh][hd][n])
  const int sec = n0 >> 9;                 // 0=q 1=k 2=v (uniform per block)
  const int h = (n0 & 511) >> 6;           // head (64-wide tile lies in one head)
  const int token0 = m0 + 16 * wave + 8 * hi;
  const int bidx = token0 >> 11;
  const int bh = bidx * 8 + h;
#pragma unroll
  for (int nt = 0; nt < 4; ++nt) {
    const int hd = nt * 16 + lo;
    const float bv = bias[n0 + hd];
    if (sec == 2) {
      v8bf pk;
#pragma unroll
      for (int r = 0; r < 8; ++r) pk[r] = (__bf16)(acc[nt][r] + bv);
      *(v8bf*)(vt + ((size_t)bh * 64 + hd) * 2048 + (token0 & 2047)) = pk;
    } else {
      __bf16* dst = (sec == 0) ? qo : ko;
#pragma unroll
      for (int r = 0; r < 8; ++r)
        dst[((size_t)bh * 2048 + ((token0 + r) & 2047)) * 64 + hd] = (__bf16)(acc[nt][r] + bv);
    }
  }
}

// ---------------------------------------------------------------------------
// Kernel 2: sliding-window flash attention, one 16-query tile per wave.
// Scores transposed: T[j][i] = K_j . Q_i (lane = query col) so softmax state
// is per-lane; output transposed: Ot[d][i] = sum_j Vt[d][j] * P[j][i].
// ---------------------------------------------------------------------------
__global__ void swa_attn_kernel(const __bf16* __restrict__ q,
                                const __bf16* __restrict__ k,
                                const __bf16* __restrict__ vt,
                                __bf16* __restrict__ ctx) {
  const int lane = threadIdx.x & 31;
  const int wave = threadIdx.x >> 5;
  const int gw = blockIdx.x * 4 + wave;   // 2048 waves
  const int qt = gw & 127;
  const int bh = gw >> 7;
  const int hi = lane >> 4, lo = lane & 15;
  const int q0 = qt * 16;
  const int i = q0 + lo;

  const __bf16* qbase = q + (size_t)bh * 2048 * 64;
  const __bf16* kbase = k + (size_t)bh * 2048 * 64;
  const __bf16* vbase = vt + (size_t)bh * 64 * 2048;

  const __bf16* qrow = qbase + (size_t)i * 64;
  const v16bf qb0 = load_a_frag(qrow, 0, hi);      // hd 0..31
  const v16bf qb1 = load_a_frag(qrow, 32, hi);     // hd 32..63

  v8f o[4] = {};
  float m_run = -1e30f, l_run = 0.f;

  for (int c = 0; c < 5; ++c) {
    const int kc = q0 - 128 + 32 * c;   // 32-key chunk

    int j0c = min(max(kc + lo, 0), 2047);
    int j1c = min(max(kc + 16 + lo, 0), 2047);
    const __bf16* kr0 = kbase + (size_t)j0c * 64;
    const __bf16* kr1 = kbase + (size_t)j1c * 64;
    v8f t0 = {}, t1 = {};
    t0 = WMMA_BF16(load_a_frag(kr0, 32, hi), qb1, t0);
    t0 = WMMA_BF16(load_a_frag(kr0, 0, hi), qb0, t0);
    t1 = WMMA_BF16(load_a_frag(kr1, 32, hi), qb1, t1);
    t1 = WMMA_BF16(load_a_frag(kr1, 0, hi), qb0, t1);

    float p0[8], p1[8];
    float mc = -1e30f;
#pragma unroll
    for (int r = 0; r < 8; ++r) {
      int jj0 = kc + r + 8 * hi;
      int jj1 = jj0 + 16;
      float s0 = (jj0 >= 0 && jj0 <= i && (i - jj0) <= 128) ? t0[r] * 0.125f : -1e30f;
      float s1 = (jj1 >= 0 && jj1 <= i && (i - jj1) <= 128) ? t1[r] * 0.125f : -1e30f;
      p0[r] = s0; p1[r] = s1;
      mc = fmaxf(mc, fmaxf(s0, s1));
    }
    mc = fmaxf(mc, __shfl_xor(mc, 16, 32));
    const float m_new = fmaxf(m_run, mc);
    const float rescale = (m_run > -1e29f) ? __expf(m_run - m_new) : 0.f;

    float lsum = 0.f;
#pragma unroll
    for (int r = 0; r < 8; ++r) {
      p0[r] = (p0[r] > -1e29f) ? __expf(p0[r] - m_new) : 0.f;
      p1[r] = (p1[r] > -1e29f) ? __expf(p1[r] - m_new) : 0.f;
      lsum += p0[r] + p1[r];
    }
    lsum += __shfl_xor(lsum, 16, 32);
    l_run = l_run * rescale + lsum;
    m_run = m_new;
#pragma unroll
    for (int hdt = 0; hdt < 4; ++hdt)
#pragma unroll
      for (int r = 0; r < 8; ++r) o[hdt][r] *= rescale;

    // P^T B-fragment: element e -> key j = kc + 16*hi + e (one shfl per VGPR)
    v16bf pb;
#pragma unroll
    for (int r = 0; r < 8; ++r) {
      float pp0 = __shfl_xor(p0[r], 16, 32);
      float pp1 = __shfl_xor(p1[r], 16, 32);
      pb[r]     = (__bf16)(hi ? pp1 : p0[r]);
      pb[8 + r] = (__bf16)(hi ? p1[r] : pp0);
    }

    int clo = min(max(kc + 16 * hi, 0), 2040);
    int chi2 = min(max(kc + 16 * hi + 8, 0), 2040);
#pragma unroll
    for (int hdt = 0; hdt < 4; ++hdt) {
      const __bf16* vrow = vbase + (size_t)(hdt * 16 + lo) * 2048;
      v8bf vl = *(const v8bf*)(vrow + clo);
      v8bf vh = *(const v8bf*)(vrow + chi2);
      v16bf va = __builtin_shufflevector(vl, vh, 0, 1, 2, 3, 4, 5, 6, 7,
                                         8, 9, 10, 11, 12, 13, 14, 15);
      o[hdt] = WMMA_BF16(va, pb, o[hdt]);
    }
  }

  const float inv_l = 1.0f / l_run;
  const int b = bh >> 3, h = bh & 7;
  __bf16* crow = ctx + ((size_t)(b * 2048 + i)) * 512 + h * 64;
#pragma unroll
  for (int hdt = 0; hdt < 4; ++hdt) {
    v8bf pk;
#pragma unroll
    for (int r = 0; r < 8; ++r) pk[r] = (__bf16)(o[hdt][r] * inv_l);
    *(v8bf*)(crow + hdt * 16 + 8 * hi) = pk;
  }
}

// ---------------------------------------------------------------------------
// Kernel 3: output projection, TDM-staged like kernel 1.  fp32 output.
// ---------------------------------------------------------------------------
__global__ void swa_out_gemm_kernel(const __bf16* __restrict__ ctx,
                                    const __bf16* __restrict__ wo,
                                    const float* __restrict__ bias,
                                    float* __restrict__ out) {
  __shared__ __bf16 smem[2][2][64 * LDS_STRIDE];
  const int lane = threadIdx.x & 31;
  const int wave = threadIdx.x >> 5;
  const int hi = lane >> 4, lo = lane & 15;
  const int tmb = blockIdx.x >> 3;   // 64 row-tiles
  const int tnb = blockIdx.x & 7;    // 8 col-tiles
  const int m0 = tmb * 64;
  const int n0 = tnb * 64;

  v8f acc[4] = {};

  if (wave == 0) {
    tdm_tile_load(ctx, 4096u, (unsigned)m0, 0u, lds_off32(&smem[0][0][0]));
    tdm_tile_load(wo, 512u, (unsigned)n0, 0u, lds_off32(&smem[0][1][0]));
  }
  for (int kk = 0; kk < 8; ++kk) {
    const int cur = kk & 1;
    if (wave == 0) {
      if (kk < 7) {
        tdm_tile_load(ctx, 4096u, (unsigned)m0, (unsigned)((kk + 1) * 64),
                      lds_off32(&smem[1 - cur][0][0]));
        tdm_tile_load(wo, 512u, (unsigned)n0, (unsigned)((kk + 1) * 64),
                      lds_off32(&smem[1 - cur][1][0]));
        __builtin_amdgcn_s_wait_tensorcnt(2);
      } else {
        __builtin_amdgcn_s_wait_tensorcnt(0);
      }
    }
    __syncthreads();
    const __bf16* arow = &smem[cur][0][(16 * wave + lo) * LDS_STRIDE];
    const __bf16* btile = &smem[cur][1][0];
#pragma unroll
    for (int kb2 = 0; kb2 < 2; ++kb2) {
      v16bf a = load16(arow + kb2 * 32 + 16 * hi);
#pragma unroll
      for (int nt = 0; nt < 4; ++nt) {
        v16bf b = load16(btile + (nt * 16 + lo) * LDS_STRIDE + kb2 * 32 + 16 * hi);
        acc[nt] = WMMA_BF16(a, b, acc[nt]);
      }
    }
    if (kk < 7) __syncthreads();
  }

#pragma unroll
  for (int nt = 0; nt < 4; ++nt) {
    const int n = n0 + nt * 16 + lo;
    const float bv = bias[n];
#pragma unroll
    for (int r = 0; r < 8; ++r) {
      int m = m0 + 16 * wave + r + 8 * hi;
      out[(size_t)m * 512 + n] = acc[nt][r] + bv;
    }
  }
}

// ---------------------------------------------------------------------------
extern "C" void kernel_launch(void* const* d_in, const int* in_sizes, int n_in,
                              void* d_out, int out_size, void* d_ws, size_t ws_size,
                              hipStream_t stream) {
  (void)in_sizes; (void)n_in; (void)out_size; (void)ws_size;
  const float* x     = (const float*)d_in[0];   // (2,2048,512)
  const float* in_w  = (const float*)d_in[1];   // (1536,512)
  const float* in_b  = (const float*)d_in[2];   // (1536,)
  const float* out_w = (const float*)d_in[3];   // (512,512)
  const float* out_b = (const float*)d_in[4];   // (512,)
  float* out = (float*)d_out;

  char* ws = (char*)d_ws;
  size_t off = 0;
  auto carve = [&](size_t elems) -> __bf16* {
    __bf16* p = (__bf16*)(ws + off);
    off = (off + elems * sizeof(__bf16) + 255) & ~(size_t)255;
    return p;
  };
  const size_t XB = 4096u * 512u;
  const size_t WQ = 1536u * 512u;
  const size_t WO = 512u * 512u;
  const size_t QK = 16u * 2048u * 64u;
  __bf16* xb  = carve(XB);
  __bf16* wqb = carve(WQ);
  __bf16* wob = carve(WO);
  __bf16* qb  = carve(QK);
  __bf16* kb  = carve(QK);
  __bf16* vtb = carve(QK);
  __bf16* ctxb = carve(XB);

  swa_cvt_bf16_kernel<<<(int)(XB / 4 + 255) / 256, 256, 0, stream>>>(x, xb, (int)(XB / 4));
  swa_cvt_bf16_kernel<<<(int)(WQ / 4 + 255) / 256, 256, 0, stream>>>(in_w, wqb, (int)(WQ / 4));
  swa_cvt_bf16_kernel<<<(int)(WO / 4 + 255) / 256, 256, 0, stream>>>(out_w, wob, (int)(WO / 4));

  // 1) QKV projection: 64x24 macro-tiles, 4 waves/block, TDM double-buffered
  swa_qkv_gemm_kernel<<<64 * 24, 128, 0, stream>>>(xb, wqb, in_b, qb, kb, vtb);

  // 2) sliding-window flash attention: 2048 waves
  swa_attn_kernel<<<512, 128, 0, stream>>>(qb, kb, vtb, ctxb);

  // 3) output projection: 64x8 macro-tiles
  swa_out_gemm_kernel<<<64 * 8, 128, 0, stream>>>(ctxb, wob, out_b, out);
}
